// GQAttention_20031727468625
// MI455X (gfx1250) — matrix operand order repaired and checked by reading, not compiled, not run on hardware
//
#include <hip/hip_runtime.h>
#include <hip/hip_bf16.h>

// ---------------------------------------------------------------------------
// GQA attention block for MI455X (gfx1250, wave32, WMMA 16x16x32 f16/f32-acc)
//   x(2048,67,512) -> q/k/v proj -> RMSNorm(q,k) -> GQA attn -> @wo
// f16 compute with fp32 accumulation. All WMMA operands are staged in LDS in
// fragment-friendly order: A row-major, B transposed (T[n][k]) so every
// fragment load is two aligned ds_load_b128 (no scalar u16 gather/packing).
// Workspace holds f16 weights + q/k/v + attention output (~424 MB in d_ws).
// ---------------------------------------------------------------------------

typedef __attribute__((ext_vector_type(16))) _Float16 v16h;
typedef __attribute__((ext_vector_type(8)))  float    v8f;

#define NB    2048
#define SEQ   67
#define HID   512
#define NH    8
#define NKVH  4
#define HD    64
#define NTOK  (NB * SEQ)          // 137216 = 1072 * 128

// ---- WMMA helpers (CDNA5 VGPR layouts, wave32) ----------------------------

__device__ __forceinline__ v8f wmma16(v16h a, v16h b, v8f c) {
  // D = A(16x32 f16) * B(32x16 f16) + C(16x16 f32)
  return __builtin_amdgcn_wmma_f32_16x16x32_f16(
      /*neg_a=*/false, a, /*neg_b=*/false, b,
      /*c_mod=*/(short)0, c, /*reuse_a=*/false, /*reuse_b=*/false);
}

// A fragment from row-major A[m][k] tile (p -> tile origin).
// lanes 0-15: row M=ln, K=0..15 ; lanes 16-31: K=16..31.
// Per-lane: 16 contiguous halves -> 2x ds_load_b128 when aligned.
__device__ __forceinline__ v16h frag_a(const _Float16* p, int ld, int ln, int hi) {
  v16h a;
#pragma unroll
  for (int e = 0; e < 16; ++e) {
    int kk = (hi << 4) + e;
    a[e] = p[ln * ld + kk];
  }
  return a;
}

// B fragment from transposed storage T[n][k] (need B[k][n] = T[n][k]).
// lanes 0-15 K=0..15, lanes 16-31 K=16..31, N=ln.
// Per-lane: 16 contiguous halves -> 2x ds_load_b128 when aligned.
__device__ __forceinline__ v16h frag_bt(const _Float16* p, int ld, int ln, int hi) {
  v16h b;
#pragma unroll
  for (int e = 0; e < 16; ++e) {
    int kk = (hi << 4) + e;
    b[e] = p[ln * ld + kk];
  }
  return b;
}

// ---- Kernel 0: f32 -> f16 weight conversion -------------------------------

__global__ __launch_bounds__(256) void cvt_f16_kernel(const float* __restrict__ in,
                                                      _Float16* __restrict__ out, int n) {
  int i = blockIdx.x * 256 + threadIdx.x;
  if (i < n) out[i] = (_Float16)in[i];
}

// ---- Kernel 1: QKV projection + per-head RMSNorm --------------------------
// Block tile: 128 tokens x 64 output cols (= exactly one head) so the
// RMSNorm reduction over D=64 is block-local. 8 waves, each 2x2 WMMA tiles,
// K chunk of 64 -> 8 WMMA between barriers.

#define TM   128
#define TN   64
#define KCH  64
#define LDA  72    // halves; 144 B row stride (16B aligned)
#define LDBT 72    // halves; Bt[n][k] row stride

__global__ __launch_bounds__(256) void qkv_rms_kernel(
    const float* __restrict__ x,
    const _Float16* __restrict__ wq_h, const _Float16* __restrict__ wk_h,
    const _Float16* __restrict__ wv_h,
    const float* __restrict__ qnw, const float* __restrict__ knw,
    _Float16* __restrict__ q_h, _Float16* __restrict__ k_h,
    _Float16* __restrict__ v_h)
{
  __shared__ _Float16 As[TM * LDA];     // A chunk, row-major [m][k]
  __shared__ _Float16 Bt[TN * LDBT];    // B chunk, transposed [n][k]
  __shared__ float    Ys[TM * TN];
  __shared__ float    Rs[TM];

  const int tid  = threadIdx.x;
  const int lane = tid & 31;
  const int wave = tid >> 5;
  const int hi   = lane >> 4;
  const int ln   = lane & 15;

  const size_t rowbase = (size_t)blockIdx.x * TM;
  const int cg = blockIdx.y;   // 0..7: q heads, 8..11: k heads, 12..15: v heads

  const _Float16* W; const float* nw; _Float16* outp;
  int ldw, colofs, ldo;
  if (cg < 8)       { W = wq_h; ldw = 512; colofs = cg * 64;        nw = qnw;     outp = q_h; ldo = 512; }
  else if (cg < 12) { W = wk_h; ldw = 256; colofs = (cg - 8) * 64;  nw = knw;     outp = k_h; ldo = 256; }
  else              { W = wv_h; ldw = 256; colofs = (cg - 12) * 64; nw = nullptr; outp = v_h; ldo = 256; }

  const int mrow = (wave >> 1) * 32;   // 0,32,64,96
  const int ncol = (wave & 1) * 32;    // 0,32

  v8f acc00 = {}, acc01 = {}, acc10 = {}, acc11 = {};

  for (int kc = 0; kc < HID; kc += KCH) {
    __syncthreads();
    // stage X chunk (f32 -> f16), coalesced global reads
    for (int i = tid; i < TM * KCH; i += 256) {
      int r = i >> 6, c = i & 63;
      As[r * LDA + c] = (_Float16)x[(rowbase + r) * HID + kc + c];
    }
    // stage W chunk transposed into Bt[n][k], coalesced global reads
    for (int i = tid; i < KCH * TN; i += 256) {
      int k = i >> 6, n = i & 63;
      Bt[n * LDBT + k] = W[(size_t)(kc + k) * ldw + colofs + n];
    }
    // prefetch next X chunk (128 rows x 256B, one cacheline-half per thread)
    if (kc + KCH < HID)
      __builtin_prefetch(&x[(rowbase + (tid >> 1)) * HID + kc + KCH + (tid & 1) * 32], 0, 3);
    __syncthreads();

#pragma unroll
    for (int kc2 = 0; kc2 < KCH; kc2 += 32) {
      v16h a0 = frag_a(&As[mrow * LDA + kc2],        LDA,  ln, hi);
      v16h a1 = frag_a(&As[(mrow + 16) * LDA + kc2], LDA,  ln, hi);
      v16h b0 = frag_bt(&Bt[ncol * LDBT + kc2],        LDBT, ln, hi);
      v16h b1 = frag_bt(&Bt[(ncol + 16) * LDBT + kc2], LDBT, ln, hi);
      acc00 = wmma16(a0, b0, acc00);
      acc01 = wmma16(a0, b1, acc01);
      acc10 = wmma16(a1, b0, acc10);
      acc11 = wmma16(a1, b1, acc11);
    }
  }

  // spill accumulators to LDS for the row-wise RMSNorm
#pragma unroll
  for (int r = 0; r < 8; ++r) {
    Ys[(mrow +      r + 8 * hi) * TN + ncol +      ln] = acc00[r];
    Ys[(mrow +      r + 8 * hi) * TN + ncol + 16 + ln] = acc01[r];
    Ys[(mrow + 16 + r + 8 * hi) * TN + ncol +      ln] = acc10[r];
    Ys[(mrow + 16 + r + 8 * hi) * TN + ncol + 16 + ln] = acc11[r];
  }
  __syncthreads();

  if (tid < TM) {
    float s = 0.f;
#pragma unroll
    for (int d = 0; d < HD; ++d) { float v = Ys[tid * TN + d]; s += v * v; }
    Rs[tid] = nw ? rsqrtf(s * (1.0f / HD) + 1e-5f) : 1.0f;
  }
  __syncthreads();

  for (int i = tid; i < TM * TN; i += 256) {
    int r = i >> 6, d = i & 63;
    float w = nw ? nw[d] : 1.0f;
    outp[(rowbase + r) * ldo + colofs + d] = (_Float16)(Ys[i] * Rs[r] * w);
  }
}

// ---- Kernel 2: GQA attention per (batch, kv-head) -------------------------
// S=67 padded to 80 rows / 96 K for exact 16x16x32 tiling; pads are zero.
// Scores fp32 in LDS; probs written f16 in place over the score buffer
// (per-row ascending byte offsets make the alias hazard-free). V is staged
// transposed (Vt[d][key]) so P.V B-fragments are contiguous LDS reads.

#define SP   80
#define SPK  96
#define LQ   64
#define LK   64
#define LVT  104       // halves; Vt[d][key] row stride (208 B, 16B aligned)
#define LS   96        // fp32 score stride
#define LP   (2 * LS)  // f16 prob stride over the aliased buffer

__global__ __launch_bounds__(256) void attn_kernel(
    const _Float16* __restrict__ q_h, const _Float16* __restrict__ k_h,
    const _Float16* __restrict__ v_h, _Float16* __restrict__ attn_h)
{
  __shared__ _Float16 Qs[SP * LQ];       // [query][d]
  __shared__ _Float16 Ks[SP * LK];       // [key][d]   (B^T for scores)
  __shared__ _Float16 Vt[HD * LVT];      // [d][key]   (B^T for P.V)
  __shared__ float    Ss[SP * LS];
  _Float16* Ps = (_Float16*)Ss;

  const int tid  = threadIdx.x;
  const int lane = tid & 31;
  const int wave = tid >> 5;
  const int hi   = lane >> 4;
  const int ln   = lane & 15;

  const int b  = blockIdx.x >> 2;
  const int kh = blockIdx.x & 3;
  const size_t tokbase = (size_t)b * SEQ;

  // stage K (80x64) and V transposed (64 x 96, zero-padded keys)
  for (int i = tid; i < SP * HD; i += 256) {
    int s = i >> 6, d = i & 63;
    _Float16 kv = (_Float16)0;
    if (s < SEQ) kv = k_h[(tokbase + s) * (NKVH * HD) + kh * HD + d];
    Ks[s * LK + d] = kv;
  }
  for (int i = tid; i < SPK * HD; i += 256) {
    int s = i >> 6, d = i & 63;
    _Float16 vv = (_Float16)0;
    if (s < SEQ) vv = v_h[(tokbase + s) * (NKVH * HD) + kh * HD + d];
    Vt[d * LVT + s] = vv;
  }

  for (int g = 0; g < 2; ++g) {        // 2 query heads per kv head
    __syncthreads();                   // also fences previous g's P.V reads
    for (int i = tid; i < SP * HD; i += 256) {
      int s = i >> 6, d = i & 63;
      _Float16 qv = (_Float16)0;
      if (s < SEQ) qv = q_h[(tokbase + s) * (NH * HD) + (kh * 2 + g) * HD + d];
      Qs[s * LQ + d] = qv;
    }
    __syncthreads();

    // scores = Q K^T * 1/sqrt(64): 5x5 WMMA tiles, K-dim = 64 (2 chunks)
    for (int t = wave; t < 25; t += 8) {
      int mt = t / 5, nt = t % 5;
      v8f acc = {};
      acc = wmma16(frag_a(&Qs[mt * 16 * LQ + 0],  LQ, ln, hi),
                   frag_bt(&Ks[nt * 16 * LK + 0],  LK, ln, hi), acc);
      acc = wmma16(frag_a(&Qs[mt * 16 * LQ + 32], LQ, ln, hi),
                   frag_bt(&Ks[nt * 16 * LK + 32], LK, ln, hi), acc);
#pragma unroll
      for (int r = 0; r < 8; ++r)
        Ss[(mt * 16 + r + 8 * hi) * LS + nt * 16 + ln] = acc[r] * 0.125f;
    }
    __syncthreads();

    // row-wise softmax over the 67 valid keys; write probs as f16
    if (tid < SP) {
      int s = tid;
      if (s < SEQ) {
        float mx = -1e30f;
        for (int c = 0; c < SEQ; ++c) mx = fmaxf(mx, Ss[s * LS + c]);
        float sum = 0.f;
        for (int c = 0; c < SEQ; ++c) {
          float e = __expf(Ss[s * LS + c] - mx);
          Ss[s * LS + c] = e;
          sum += e;
        }
        float inv = 1.0f / sum;
        for (int c = 0; c < SPK; ++c)   // ascending c: f16 write never clobbers unread f32
          Ps[s * LP + c] = (c < SEQ) ? (_Float16)(Ss[s * LS + c] * inv) : (_Float16)0;
      } else {
        for (int c = 0; c < SPK; ++c) Ps[s * LP + c] = (_Float16)0;
      }
    }
    __syncthreads();

    // out = P V: 5x4 WMMA tiles, K-dim = 96 (3 chunks, pads are zero)
    for (int t = wave; t < 20; t += 8) {
      int mt = t / 4, nt = t % 4;
      v8f acc = {};
#pragma unroll
      for (int kc = 0; kc < SPK; kc += 32)
        acc = wmma16(frag_a(&Ps[mt * 16 * LP + kc], LP, ln, hi),
                     frag_bt(&Vt[(nt * 16) * LVT + kc], LVT, ln, hi), acc);
#pragma unroll
      for (int r = 0; r < 8; ++r) {
        int row = mt * 16 + r + 8 * hi;
        if (row < SEQ)
          attn_h[(tokbase + row) * (NH * HD) + (kh * 2 + g) * HD + nt * 16 + ln] =
              (_Float16)acc[r];
      }
    }
  }
}

// ---- Kernel 3: output projection (attn @ wo), fp32 result -----------------

__global__ __launch_bounds__(256) void out_gemm_kernel(
    const _Float16* __restrict__ attn_h, const _Float16* __restrict__ wo_h,
    float* __restrict__ out)
{
  __shared__ _Float16 As[TM * LDA];
  __shared__ _Float16 Bt[TN * LDBT];

  const int tid  = threadIdx.x;
  const int lane = tid & 31;
  const int wave = tid >> 5;
  const int hi   = lane >> 4;
  const int ln   = lane & 15;

  const size_t rowbase = (size_t)blockIdx.x * TM;
  const int colbase = blockIdx.y * 64;

  const int mrow = (wave >> 1) * 32;
  const int ncol = (wave & 1) * 32;

  v8f acc00 = {}, acc01 = {}, acc10 = {}, acc11 = {};

  for (int kc = 0; kc < HID; kc += KCH) {
    __syncthreads();
    for (int i = tid; i < TM * KCH; i += 256) {
      int r = i >> 6, c = i & 63;
      As[r * LDA + c] = attn_h[(rowbase + r) * HID + kc + c];
    }
    for (int i = tid; i < KCH * TN; i += 256) {
      int k = i >> 6, n = i & 63;
      Bt[n * LDBT + k] = wo_h[(size_t)(kc + k) * HID + colbase + n];
    }
    if (kc + KCH < HID)
      __builtin_prefetch(&attn_h[(rowbase + (tid >> 1)) * HID + kc + KCH + (tid & 1) * 32], 0, 3);
    __syncthreads();

#pragma unroll
    for (int kc2 = 0; kc2 < KCH; kc2 += 32) {
      v16h a0 = frag_a(&As[mrow * LDA + kc2],        LDA,  ln, hi);
      v16h a1 = frag_a(&As[(mrow + 16) * LDA + kc2], LDA,  ln, hi);
      v16h b0 = frag_bt(&Bt[ncol * LDBT + kc2],        LDBT, ln, hi);
      v16h b1 = frag_bt(&Bt[(ncol + 16) * LDBT + kc2], LDBT, ln, hi);
      acc00 = wmma16(a0, b0, acc00);
      acc01 = wmma16(a0, b1, acc01);
      acc10 = wmma16(a1, b0, acc10);
      acc11 = wmma16(a1, b1, acc11);
    }
  }

#pragma unroll
  for (int r = 0; r < 8; ++r) {
    out[(rowbase + mrow +      r + 8 * hi) * HID + colbase + ncol +      ln] = acc00[r];
    out[(rowbase + mrow +      r + 8 * hi) * HID + colbase + ncol + 16 + ln] = acc01[r];
    out[(rowbase + mrow + 16 + r + 8 * hi) * HID + colbase + ncol +      ln] = acc10[r];
    out[(rowbase + mrow + 16 + r + 8 * hi) * HID + colbase + ncol + 16 + ln] = acc11[r];
  }
}

// ---- Launcher -------------------------------------------------------------

extern "C" void kernel_launch(void* const* d_in, const int* in_sizes, int n_in,
                              void* d_out, int out_size, void* d_ws, size_t ws_size,
                              hipStream_t stream) {
  const float* x   = (const float*)d_in[0];
  const float* wq  = (const float*)d_in[1];
  const float* wk  = (const float*)d_in[2];
  const float* wv  = (const float*)d_in[3];
  const float* wo  = (const float*)d_in[4];
  const float* qnw = (const float*)d_in[5];
  const float* knw = (const float*)d_in[6];
  float* out = (float*)d_out;

  // workspace layout (f16 elements); requires ~424 MB
  _Float16* ws     = (_Float16*)d_ws;
  _Float16* wq_h   = ws;
  _Float16* wk_h   = wq_h + (size_t)512 * 512;
  _Float16* wv_h   = wk_h + (size_t)512 * 256;
  _Float16* wo_h   = wv_h + (size_t)512 * 256;
  _Float16* q_h    = wo_h + (size_t)512 * 512;
  _Float16* k_h    = q_h  + (size_t)NTOK * 512;
  _Float16* v_h    = k_h  + (size_t)NTOK * 256;
  _Float16* attn_h = v_h  + (size_t)NTOK * 256;

  cvt_f16_kernel<<<(512 * 512 + 255) / 256, 256, 0, stream>>>(wq, wq_h, 512 * 512);
  cvt_f16_kernel<<<(512 * 256 + 255) / 256, 256, 0, stream>>>(wk, wk_h, 512 * 256);
  cvt_f16_kernel<<<(512 * 256 + 255) / 256, 256, 0, stream>>>(wv, wv_h, 512 * 256);
  cvt_f16_kernel<<<(512 * 512 + 255) / 256, 256, 0, stream>>>(wo, wo_h, 512 * 512);

  qkv_rms_kernel<<<dim3(NTOK / TM, 16), 256, 0, stream>>>(
      x, wq_h, wk_h, wv_h, qnw, knw, q_h, k_h, v_h);

  attn_kernel<<<NB * NKVH, 256, 0, stream>>>(q_h, k_h, v_h, attn_h);

  out_gemm_kernel<<<dim3(NTOK / TM, 8), 256, 0, stream>>>(attn_h, wo_h, out);
}
